// NodeEdgeModel_39135742001770
// MI455X (gfx1250) — compile-verified, matrix operand order checked by reading, not run on hardware
//
#include <hip/hip_runtime.h>

#define H_DIM   128
#define HID     256
#define NNODES  50000
#define NEDGES  800000

typedef __attribute__((ext_vector_type(16))) __bf16 v16bf;
typedef __attribute__((ext_vector_type(8)))  __bf16 v8bf;
typedef __attribute__((ext_vector_type(8)))  float  v8f;

static __device__ __forceinline__ unsigned short f32_bf16(float f) {
  unsigned int u = __float_as_uint(f);
  u += 0x7FFFu + ((u >> 16) & 1u);            // round-to-nearest-even
  return (unsigned short)(u >> 16);
}

static __device__ __forceinline__ float silu(float x) {
  return x / (1.0f + __expf(-x));
}

// B fragment: pre-packed so each lane reads 32 contiguous bytes.
static __device__ __forceinline__ v16bf load_b_frag(const unsigned short* __restrict__ pack,
                                                    int frag, int lane) {
  return *(const v16bf*)(pack + ((size_t)(frag * 32 + lane) << 4));
}

// A fragment from LDS row (row-major bf16). ISA 16-bit A 16x32 layout:
// lane<16: K = {0..7,16..23}; lane>=16: +8.  kb = 8*(lane>>4).
static __device__ __forceinline__ v16bf load_a_frag(const unsigned short* rowp, int ks, int kb) {
  v8bf lo = *(const v8bf*)(rowp + ks * 32 + kb);
  v8bf hi = *(const v8bf*)(rowp + ks * 32 + 16 + kb);
  return __builtin_shufflevector(lo, hi, 0,1,2,3,4,5,6,7,8,9,10,11,12,13,14,15);
}

static __device__ __forceinline__ void atomic_add_f32(float* p, float v) {
  asm volatile("global_atomic_add_f32 %0, %1, off" :: "v"(p), "v"(v) : "memory");
}

// Pack W[OUT,IN] -> WMMA B fragments (B(k,n)=W[n,k]) in bf16, lane-contiguous.
__global__ void pack_w(const float* __restrict__ W, int IN, int NT,
                       unsigned short* __restrict__ dst, int total) {
  int t = blockIdx.x * 256 + threadIdx.x;
  if (t >= total) return;
  int f = t >> 9, r = t & 511;
  int lane = r >> 4, i = r & 15;
  int ks = f / NT, nt = f - ks * NT;
  int n = nt * 16 + (lane & 15);
  int k = ks * 32 + ((lane >> 4) << 3) + (i & 7) + ((i >= 8) ? 16 : 0);
  dst[t] = f32_bf16(W[(size_t)n * IN + k]);
}

__global__ void prep_wrad(const float* __restrict__ We1, float* __restrict__ wrad) {
  int t = threadIdx.x;                        // 256 threads
  wrad[t] = We1[(size_t)t * 257 + 256];       // the radial column, kept fp32-exact
}

// ---------------- fused edge MLP + scatter-add ----------------
__global__ __launch_bounds__(128) void edge_kernel(
    const float* __restrict__ h, const float* __restrict__ cd,
    const long long* __restrict__ row, const long long* __restrict__ col,
    const unsigned short* __restrict__ pWe1, const unsigned short* __restrict__ pWe2,
    const float* __restrict__ wrad, const float* __restrict__ be1,
    const float* __restrict__ be2, float* __restrict__ agg) {
  __shared__ unsigned short sAct[4][16][264];   // 264 = 256 + 8 pad (bank stride 4)
  __shared__ float sRad[4][16];
  __shared__ int   sRow[4][16];

  const int tid = threadIdx.x, w = tid >> 5, lane = tid & 31;
  const int n0 = lane & 15, hi = lane >> 4, kb = hi << 3;
  const int e0 = blockIdx.x * 64 + w * 16;

  if (lane < 16) {
    int e = e0 + lane;
    float x = cd[3 * e], y = cd[3 * e + 1], z = cd[3 * e + 2];
    sRad[w][lane] = x * x + y * y + z * z;
    sRow[w][lane] = (int)row[e];
  }

  // Stage gathered h[row]||h[col] as bf16 (coalesced b128 loads).
  for (int m = 0; m < 16; ++m) {
    int e = e0 + m;
    size_t nr = (size_t)row[e], nc = (size_t)col[e];
    float4 a = ((const float4*)(h + nr * H_DIM))[lane];
    float4 b = ((const float4*)(h + nc * H_DIM))[lane];
    ushort4 pa = make_ushort4(f32_bf16(a.x), f32_bf16(a.y), f32_bf16(a.z), f32_bf16(a.w));
    ushort4 pb = make_ushort4(f32_bf16(b.x), f32_bf16(b.y), f32_bf16(b.z), f32_bf16(b.w));
    *(ushort4*)&sAct[w][m][lane * 4]       = pa;
    *(ushort4*)&sAct[w][m][128 + lane * 4] = pb;
  }

  float rad[8]; int nodev[8];
  #pragma unroll
  for (int v = 0; v < 8; ++v) { rad[v] = sRad[w][v + kb]; nodev[v] = sRow[w][v + kb]; }

  const unsigned short* myrow = &sAct[w][n0][0];
  v8f acc[16];

  // ---- layer 1: bias + fp32 rank-1 radial term, then 8 K-steps x 16 N-tiles of WMMA
  #pragma unroll
  for (int nt = 0; nt < 16; ++nt) {
    float b = be1[nt * 16 + n0], wr = wrad[nt * 16 + n0];
    #pragma unroll
    for (int v = 0; v < 8; ++v) acc[nt][v] = b + rad[v] * wr;
  }
  #pragma unroll
  for (int ks = 0; ks < 8; ++ks) {
    v16bf a = load_a_frag(myrow, ks, kb);
    #pragma unroll
    for (int nt = 0; nt < 16; ++nt) {
      v16bf bf = load_b_frag(pWe1, ks * 16 + nt, lane);
      acc[nt] = __builtin_amdgcn_wmma_f32_16x16x32_bf16(false, a, false, bf,
                                                        (short)0, acc[nt], false, false);
    }
  }

  // SiLU, write m tile back to (wave-private) LDS as bf16 for layer-2 A fragments.
  #pragma unroll
  for (int nt = 0; nt < 16; ++nt)
    #pragma unroll
    for (int v = 0; v < 8; ++v)
      sAct[w][v + kb][nt * 16 + n0] = f32_bf16(silu(acc[nt][v]));

  // ---- layer 2
  #pragma unroll
  for (int nt = 0; nt < 16; ++nt) {
    float b = be2[nt * 16 + n0];
    #pragma unroll
    for (int v = 0; v < 8; ++v) acc[nt][v] = b;
  }
  #pragma unroll
  for (int ks = 0; ks < 8; ++ks) {
    v16bf a = load_a_frag(myrow, ks, kb);
    #pragma unroll
    for (int nt = 0; nt < 16; ++nt) {
      v16bf bf = load_b_frag(pWe2, ks * 16 + nt, lane);
      acc[nt] = __builtin_amdgcn_wmma_f32_16x16x32_bf16(false, a, false, bf,
                                                        (short)0, acc[nt], false, false);
    }
  }

  // SiLU + scatter-add into agg (L2-resident, no-return atomics).
  #pragma unroll
  for (int nt = 0; nt < 16; ++nt)
    #pragma unroll
    for (int v = 0; v < 8; ++v)
      atomic_add_f32(&agg[(size_t)nodev[v] * HID + nt * 16 + n0], silu(acc[nt][v]));
}

// ---------------- node MLP ----------------
__global__ __launch_bounds__(128) void node_kernel(
    const float* __restrict__ h, const float* __restrict__ agg,
    const unsigned short* __restrict__ pWn1, const unsigned short* __restrict__ pWn2,
    const float* __restrict__ bn1, const float* __restrict__ bn2,
    float* __restrict__ out) {
  __shared__ unsigned short sAct[4][16][392];   // 392 = 384 + 8 pad

  const int tid = threadIdx.x, w = tid >> 5, lane = tid & 31;
  const int n0 = lane & 15, hi = lane >> 4, kb = hi << 3;
  const int nb = blockIdx.x * 64 + w * 16;

  // Stage [h(128) || agg(256)] per node as bf16.
  for (int m = 0; m < 16; ++m) {
    int node = min(nb + m, NNODES - 1);
    float4 a = ((const float4*)(h + (size_t)node * H_DIM))[lane];
    const float4* ap = (const float4*)(agg + (size_t)node * HID);
    float4 b = ap[lane];
    float4 c = ap[lane + 32];
    *(ushort4*)&sAct[w][m][lane * 4] =
        make_ushort4(f32_bf16(a.x), f32_bf16(a.y), f32_bf16(a.z), f32_bf16(a.w));
    *(ushort4*)&sAct[w][m][128 + lane * 4] =
        make_ushort4(f32_bf16(b.x), f32_bf16(b.y), f32_bf16(b.z), f32_bf16(b.w));
    *(ushort4*)&sAct[w][m][256 + lane * 4] =
        make_ushort4(f32_bf16(c.x), f32_bf16(c.y), f32_bf16(c.z), f32_bf16(c.w));
  }

  const unsigned short* myrow = &sAct[w][n0][0];
  v8f acc[16];

  // ---- layer 1: K=384 (12 K-steps), 256 outputs
  #pragma unroll
  for (int nt = 0; nt < 16; ++nt) {
    float b = bn1[nt * 16 + n0];
    #pragma unroll
    for (int v = 0; v < 8; ++v) acc[nt][v] = b;
  }
  #pragma unroll
  for (int ks = 0; ks < 12; ++ks) {
    v16bf a = load_a_frag(myrow, ks, kb);
    #pragma unroll
    for (int nt = 0; nt < 16; ++nt) {
      v16bf bf = load_b_frag(pWn1, ks * 16 + nt, lane);
      acc[nt] = __builtin_amdgcn_wmma_f32_16x16x32_bf16(false, a, false, bf,
                                                        (short)0, acc[nt], false, false);
    }
  }
  #pragma unroll
  for (int nt = 0; nt < 16; ++nt)
    #pragma unroll
    for (int v = 0; v < 8; ++v)
      sAct[w][v + kb][nt * 16 + n0] = f32_bf16(silu(acc[nt][v]));

  // ---- layer 2: K=256 (8 K-steps), 128 outputs, no activation
  v8f acc2[8];
  #pragma unroll
  for (int nt = 0; nt < 8; ++nt) {
    float b = bn2[nt * 16 + n0];
    #pragma unroll
    for (int v = 0; v < 8; ++v) acc2[nt][v] = b;
  }
  #pragma unroll
  for (int ks = 0; ks < 8; ++ks) {
    v16bf a = load_a_frag(myrow, ks, kb);
    #pragma unroll
    for (int nt = 0; nt < 8; ++nt) {
      v16bf bf = load_b_frag(pWn2, ks * 8 + nt, lane);
      acc2[nt] = __builtin_amdgcn_wmma_f32_16x16x32_bf16(false, a, false, bf,
                                                         (short)0, acc2[nt], false, false);
    }
  }
  #pragma unroll
  for (int nt = 0; nt < 8; ++nt)
    #pragma unroll
    for (int v = 0; v < 8; ++v) {
      int node = nb + v + kb;
      if (node < NNODES) out[(size_t)node * H_DIM + nt * 16 + n0] = acc2[nt][v];
    }
}

extern "C" void kernel_launch(void* const* d_in, const int* in_sizes, int n_in,
                              void* d_out, int out_size, void* d_ws, size_t ws_size,
                              hipStream_t stream) {
  const float*     h   = (const float*)d_in[0];
  const float*     cd  = (const float*)d_in[1];
  const long long* row = (const long long*)d_in[2];
  const long long* col = (const long long*)d_in[3];
  const float*     We1 = (const float*)d_in[4];
  const float*     be1 = (const float*)d_in[5];
  const float*     We2 = (const float*)d_in[6];
  const float*     be2 = (const float*)d_in[7];
  const float*     Wn1 = (const float*)d_in[8];
  const float*     bn1 = (const float*)d_in[9];
  const float*     Wn2 = (const float*)d_in[10];
  const float*     bn2 = (const float*)d_in[11];
  float* out = (float*)d_out;

  char* ws = (char*)d_ws;
  size_t off = 0;
  float* agg = (float*)(ws + off);            off += (size_t)NNODES * HID * 4;
  unsigned short* pWe1 = (unsigned short*)(ws + off); off += 256 * 256 * 2;
  unsigned short* pWe2 = (unsigned short*)(ws + off); off += 256 * 256 * 2;
  unsigned short* pWn1 = (unsigned short*)(ws + off); off += 384 * 256 * 2;
  unsigned short* pWn2 = (unsigned short*)(ws + off); off += 256 * 128 * 2;
  float* wrad = (float*)(ws + off);           off += 256 * 4;

  hipMemsetAsync(agg, 0, (size_t)NNODES * HID * 4, stream);
  pack_w<<<(16 * 8 * 512) / 256, 256, 0, stream>>>(We1, 257, 16, pWe1, 16 * 8 * 512);
  pack_w<<<(16 * 8 * 512) / 256, 256, 0, stream>>>(We2, 256, 16, pWe2, 16 * 8 * 512);
  pack_w<<<(16 * 12 * 512) / 256, 256, 0, stream>>>(Wn1, 384, 16, pWn1, 16 * 12 * 512);
  pack_w<<<(8 * 8 * 512) / 256, 256, 0, stream>>>(Wn2, 256, 8, pWn2, 8 * 8 * 512);
  prep_wrad<<<1, 256, 0, stream>>>(We1, wrad);

  edge_kernel<<<NEDGES / 64, 128, 0, stream>>>(h, cd, row, col, pWe1, pWe2, wrad, be1, be2, agg);
  node_kernel<<<(NNODES + 63) / 64, 128, 0, stream>>>(h, agg, pWn1, pWn2, bn1, bn2, out);
}
